// Net_2817498546281
// MI455X (gfx1250) — compile-verified, matrix operand order checked by reading, not run on hardware
//
#include <hip/hip_runtime.h>
#include <hip/hip_bf16.h>

#define NND   20000
#define DIMD  256
#define IND   1000
#define INP   1024
#define ED    320000
#define BD    128
#define CLSD  121

#define USE_ASYNC_LDS 1

typedef __attribute__((ext_vector_type(16))) _Float16 v16h;
typedef __attribute__((ext_vector_type(8)))  float    v8f;

// ---------------- conversion kernels ----------------

__global__ void cvt_x_kernel(const float* __restrict__ x, _Float16* __restrict__ x16, int elems) {
  int i = blockIdx.x * blockDim.x + threadIdx.x;
  if (i >= elems) return;
  int n = i >> 10, k = i & 1023;                       // padded to 1024 cols
  x16[i] = (k < IND) ? (_Float16)x[n * IND + k] : (_Float16)0.f;
}

// W [K, Nc] f32 row-major  ->  Wt [Npad, Kpad] f16 (transposed, zero-padded)
__global__ void cvt_wt_kernel(const float* __restrict__ W, _Float16* __restrict__ Wt,
                              int K, int Nc, int Kpad, int elems) {
  int i = blockIdx.x * blockDim.x + threadIdx.x;
  if (i >= elems) return;
  int n = i / Kpad, k = i - n * Kpad;
  Wt[i] = (n < Nc && k < K) ? (_Float16)W[(size_t)k * Nc + n] : (_Float16)0.f;
}

// ---------------- WMMA GEMM: C[M,N] = A[M,K](f16) * Wt[N,K](f16)^T ----------------
// 256 threads = 8 waves, arranged 2 (M) x 4 (N); block tile 32 x 128,
// wave tile 16 x 32 (two 16x16 accumulators sharing one A fragment).
// A tile (32x32 f16 = 2KB) staged in LDS via global_load_async_to_lds_b128.
__global__ __launch_bounds__(256)
void gemm_wmma_kernel(const _Float16* __restrict__ A, int lda,
                      const _Float16* __restrict__ Bt, int ldb,
                      const float* __restrict__ bias, int do_relu,
                      _Float16* __restrict__ C16, int ldc16, int coloff,
                      float* __restrict__ C32, int ldc32,
                      int Nvalid, int K)
{
  __shared__ _Float16 tileA[32 * 32];

  const int tid  = threadIdx.x;
  const int wave = tid >> 5;
  const int lane = tid & 31;
  const int mgrp = wave >> 2;            // 0/1: which 16-row half of the block tile
  const int ngrp = wave & 3;             // 0..3: which 32-col group
  const int m0 = blockIdx.x * 32;
  const int n0 = blockIdx.y * 128 + ngrp * 32;
  const int mm = lane & 15;
  const int lh = lane >> 4;              // lane half selects K sub-range

  // staging assignment: threads 0..127 each move 16 bytes of the 32x32 A tile
  const int srow = tid >> 2;             // 0..31
  const int sseg = (tid & 3) * 8;        // halves offset within row: 0,8,16,24
  const _Float16* __restrict__ Ag = A + (size_t)(m0 + srow) * lda + sseg;
  _Float16* ldsPtr = tileA + srow * 32 + sseg;
  const unsigned ldsDst = (unsigned)(size_t)ldsPtr;

  const _Float16* __restrict__ Brow0 = Bt + (size_t)(n0 + mm) * ldb;
  const _Float16* __restrict__ Brow1 = Bt + (size_t)(n0 + 16 + mm) * ldb;
  const _Float16* lA = tileA + (mgrp * 16 + mm) * 32;

  v8f acc0 = {}, acc1 = {};
  for (int k0 = 0; k0 < K; k0 += 32) {
    if (tid < 128) {                     // waves 0..3 stage the A tile
#if USE_ASYNC_LDS
      unsigned long long ga = (unsigned long long)(const void*)(Ag + k0);
      asm volatile("global_load_async_to_lds_b128 %0, %1, off"
                   :: "v"(ldsDst), "v"(ga) : "memory");
#else
      *(uint4*)ldsPtr = *(const uint4*)(Ag + k0);
#endif
    }
#if USE_ASYNC_LDS
    asm volatile("s_wait_asynccnt 0x0" ::: "memory");
#endif
    __syncthreads();                     // publish LDS tile to all 8 waves

    union { v16h v; uint4 q[2]; } a, b0, b1;
    // ISA 7.12.2 f16 A-fragment: halves {lh*8..+7} then {lh*8+16..+23} -> two 16B loads
    a.q[0]  = *(const uint4*)(lA + lh * 8);
    a.q[1]  = *(const uint4*)(lA + lh * 8 + 16);
    b0.q[0] = *(const uint4*)(Brow0 + k0 + lh * 8);
    b0.q[1] = *(const uint4*)(Brow0 + k0 + lh * 8 + 16);
    b1.q[0] = *(const uint4*)(Brow1 + k0 + lh * 8);
    b1.q[1] = *(const uint4*)(Brow1 + k0 + lh * 8 + 16);
    if (k0 + 32 < K) {
      __builtin_prefetch(Brow0 + k0 + 32, 0, 3);
      __builtin_prefetch(Brow1 + k0 + 32, 0, 3);
    }
    acc0 = __builtin_amdgcn_wmma_f32_16x16x32_f16(false, a.v, false, b0.v,
                                                  (short)0, acc0, false, false);
    acc1 = __builtin_amdgcn_wmma_f32_16x16x32_f16(false, a.v, false, b1.v,
                                                  (short)0, acc1, false, false);
    __syncthreads();                     // tile reuse guard
  }

  // C layout: acc[r] -> row m0 + mgrp*16 + r + lh*8, cols n0+mm and n0+16+mm
  const int nA = n0 + mm, nB = n0 + 16 + mm;
  const float bvA = (bias && nA < Nvalid) ? bias[nA] : 0.f;
  const float bvB = (bias && nB < Nvalid) ? bias[nB] : 0.f;
#pragma unroll
  for (int r = 0; r < 8; ++r) {
    const int m = m0 + mgrp * 16 + r + lh * 8;
    if (nA < Nvalid) {
      float t = acc0[r] + bvA;
      if (do_relu) t = fmaxf(t, 0.f);
      if (C16) C16[(size_t)m * ldc16 + coloff + nA] = (_Float16)t;
      if (C32) C32[(size_t)m * ldc32 + nA] = t;
    }
    if (nB < Nvalid) {
      float t = acc1[r] + bvB;
      if (do_relu) t = fmaxf(t, 0.f);
      if (C16) C16[(size_t)m * ldc16 + coloff + nB] = (_Float16)t;
      if (C32) C32[(size_t)m * ldc32 + nB] = t;
    }
  }
}

// ---------------- edge scatter-add (post-linear aggregation, 256-dim f32) ----------------
__global__ void scatter_add_kernel(const float* __restrict__ y,
                                   const int* __restrict__ ei, int E,
                                   float* __restrict__ agg) {
  long long t = (long long)blockIdx.x * blockDim.x + threadIdx.x;
  int e = (int)(t >> 6);
  if (e >= E) return;
  int c = ((int)t & 63) * 4;
  int s = ei[e], d = ei[E + e];
  float4 val = *(const float4*)(y + (size_t)s * DIMD + c);
  float* out = agg + (size_t)d * DIMD + c;
  atomicAdd(out + 0, val.x);
  atomicAdd(out + 1, val.y);
  atomicAdd(out + 2, val.z);
  atomicAdd(out + 3, val.w);
}

// t16 = f16( relu( BN( y + agg + b ) ) )
__global__ void bnrelu_kernel(const float* __restrict__ y, const float* __restrict__ agg,
                              const float* __restrict__ b, const float* __restrict__ mn,
                              const float* __restrict__ vr, const float* __restrict__ g,
                              const float* __restrict__ be,
                              _Float16* __restrict__ out, int elems) {
  int i = blockIdx.x * blockDim.x + threadIdx.x;
  if (i >= elems) return;
  int d = i & (DIMD - 1);
  float t = y[i] + agg[i] + b[d];
  t = (t - mn[d]) * rsqrtf(vr[d] + 1e-5f) * g[d] + be[d];
  out[i] = (_Float16)fmaxf(t, 0.f);
}

// ---------------- pooling ----------------
__global__ void pool_cnt_kernel(const int* __restrict__ i1, const int* __restrict__ i2,
                                float* __restrict__ cnt, int Nn) {
  int n = blockIdx.x * blockDim.x + threadIdx.x;
  if (n >= Nn) return;
  atomicAdd(&cnt[i1[n]], 1.f);
  atomicAdd(&cnt[BD + i2[n]], 1.f);
}

__global__ void pool_sum_kernel(const float* __restrict__ h,
                                const int* __restrict__ i1, const int* __restrict__ i2,
                                float* __restrict__ gsum, int Nn) {
  long long t = (long long)blockIdx.x * blockDim.x + threadIdx.x;
  int n = (int)(t >> 6);
  if (n >= Nn) return;
  int c = ((int)t & 63) * 4;
  float4 v = *(const float4*)(h + (size_t)n * DIMD + c);
  int b1 = i1[n], b2 = i2[n];
  float* o1 = gsum + (size_t)b1 * 512 + c;
  float* o2 = gsum + (size_t)b2 * 512 + 256 + c;
  atomicAdd(o1 + 0, v.x); atomicAdd(o1 + 1, v.y); atomicAdd(o1 + 2, v.z); atomicAdd(o1 + 3, v.w);
  atomicAdd(o2 + 0, v.x); atomicAdd(o2 + 1, v.y); atomicAdd(o2 + 2, v.z); atomicAdd(o2 + 3, v.w);
}

__global__ void pool_div_kernel(const float* __restrict__ gsum, const float* __restrict__ cnt,
                                _Float16* __restrict__ g16, int elems) {
  int i = blockIdx.x * blockDim.x + threadIdx.x;
  if (i >= elems) return;
  int b = i >> 9, c = i & 511;
  float ct = fmaxf(cnt[(c < 256 ? 0 : BD) + b], 1.f);
  g16[i] = (_Float16)(gsum[i] / ct);
}

// ---------------- log-softmax over [128, 121] ----------------
__global__ void logsoftmax_kernel(const float* __restrict__ lg, float* __restrict__ out) {
  int r = threadIdx.x;
  if (r >= BD) return;
  const float* row = lg + (size_t)r * 128;
  float mx = -1e30f;
  for (int c = 0; c < CLSD; ++c) mx = fmaxf(mx, row[c]);
  float s = 0.f;
  for (int c = 0; c < CLSD; ++c) s += __expf(row[c] - mx);
  float ls = __logf(s);
  for (int c = 0; c < CLSD; ++c) out[(size_t)r * CLSD + c] = row[c] - mx - ls;
}

// ---------------- host ----------------

static inline int ceil_div(long long a, int b) { return (int)((a + b - 1) / b); }

extern "C" void kernel_launch(void* const* d_in, const int* in_sizes, int n_in,
                              void* d_out, int out_size, void* d_ws, size_t ws_size,
                              hipStream_t stream) {
  const float* x    = (const float*)d_in[0];
  const int*   e1   = (const int*)d_in[1];
  const int*   e2   = (const int*)d_in[2];
  const int*   idx1 = (const int*)d_in[3];
  const int*   idx2 = (const int*)d_in[4];
  auto prm = [&](int i) { return (const float*)d_in[i]; };

  // absolute param bases: conv = {l1.w,l1.b,bn.g,bn.be,bn.m,bn.v,l2.w,l2.b}; mlp = {l1.w,l1.b,l2.w,l2.b}
  const int C11 = 5, C12 = 13, M1 = 21, C21 = 25, C22 = 33, M2 = 41, C31 = 45, C32_ = 53, MO = 61;

  char* ws = (char*)d_ws;
  size_t off = 0;
  auto alloc = [&](size_t bytes) -> void* {
    void* p = ws + off;
    off = (off + bytes + 255) & ~(size_t)255;
    return p;
  };

  _Float16* x16  = (_Float16*)alloc((size_t)NND * INP * 2);
  _Float16* w11a = (_Float16*)alloc((size_t)DIMD * INP * 2);   // conv_1_1 l1^T [256,1024]
  _Float16* w12a = (_Float16*)alloc((size_t)DIMD * INP * 2);
  _Float16* w11b = (_Float16*)alloc((size_t)DIMD * DIMD * 2);  // conv l2^T [256,256]
  _Float16* w12b = (_Float16*)alloc((size_t)DIMD * DIMD * 2);
  _Float16* w21a = (_Float16*)alloc((size_t)DIMD * DIMD * 2);
  _Float16* w21b = (_Float16*)alloc((size_t)DIMD * DIMD * 2);
  _Float16* w22a = (_Float16*)alloc((size_t)DIMD * DIMD * 2);
  _Float16* w22b = (_Float16*)alloc((size_t)DIMD * DIMD * 2);
  _Float16* w31a = (_Float16*)alloc((size_t)DIMD * DIMD * 2);
  _Float16* w31b = (_Float16*)alloc((size_t)DIMD * DIMD * 2);
  _Float16* w32a = (_Float16*)alloc((size_t)DIMD * DIMD * 2);
  _Float16* w32b = (_Float16*)alloc((size_t)DIMD * DIMD * 2);
  _Float16* wm1a = (_Float16*)alloc((size_t)DIMD * 512 * 2);   // mlp l1^T [256,512]
  _Float16* wm1b = (_Float16*)alloc((size_t)DIMD * DIMD * 2);
  _Float16* wm2a = (_Float16*)alloc((size_t)DIMD * 512 * 2);
  _Float16* wm2b = (_Float16*)alloc((size_t)DIMD * DIMD * 2);
  _Float16* wmoa = (_Float16*)alloc((size_t)DIMD * 512 * 2);
  _Float16* wmob = (_Float16*)alloc((size_t)128 * DIMD * 2);   // [128(pad 121), 256]

  float*    y      = (float*)alloc((size_t)NND * DIMD * 4);
  float*    agg    = (float*)alloc((size_t)NND * DIMD * 4);
  _Float16* t16    = (_Float16*)alloc((size_t)NND * DIMD * 2);
  _Float16* hcat16 = (_Float16*)alloc((size_t)NND * 512 * 2);
  _Float16* h16    = (_Float16*)alloc((size_t)NND * DIMD * 2);
  float*    hf32   = (float*)alloc((size_t)NND * DIMD * 4);
  float*    gsum   = (float*)alloc((size_t)BD * 512 * 4);
  float*    cnt    = (float*)alloc((size_t)2 * BD * 4);
  _Float16* g16    = (_Float16*)alloc((size_t)BD * 512 * 2);
  _Float16* p16    = (_Float16*)alloc((size_t)BD * DIMD * 2);
  float*    lgts   = (float*)alloc((size_t)BD * 128 * 4);

  const int TB = 256;

  // --- convert inputs/weights to f16 (transposed, padded) ---
  {
    int elems = NND * INP;
    cvt_x_kernel<<<ceil_div(elems, TB), TB, 0, stream>>>(x, x16, elems);
  }
  auto cvtw = [&](const float* W, _Float16* Wt, int K, int Nc, int Kpad, int Npad) {
    int elems = Npad * Kpad;
    cvt_wt_kernel<<<ceil_div(elems, TB), TB, 0, stream>>>(W, Wt, K, Nc, Kpad, elems);
  };
  cvtw(prm(C11 + 0), w11a, IND, DIMD, INP, DIMD);  cvtw(prm(C11 + 6), w11b, DIMD, DIMD, DIMD, DIMD);
  cvtw(prm(C12 + 0), w12a, IND, DIMD, INP, DIMD);  cvtw(prm(C12 + 6), w12b, DIMD, DIMD, DIMD, DIMD);
  cvtw(prm(C21 + 0), w21a, DIMD, DIMD, DIMD, DIMD); cvtw(prm(C21 + 6), w21b, DIMD, DIMD, DIMD, DIMD);
  cvtw(prm(C22 + 0), w22a, DIMD, DIMD, DIMD, DIMD); cvtw(prm(C22 + 6), w22b, DIMD, DIMD, DIMD, DIMD);
  cvtw(prm(C31 + 0), w31a, DIMD, DIMD, DIMD, DIMD); cvtw(prm(C31 + 6), w31b, DIMD, DIMD, DIMD, DIMD);
  cvtw(prm(C32_ + 0), w32a, DIMD, DIMD, DIMD, DIMD); cvtw(prm(C32_ + 6), w32b, DIMD, DIMD, DIMD, DIMD);
  cvtw(prm(M1 + 0), wm1a, 512, DIMD, 512, DIMD);   cvtw(prm(M1 + 2), wm1b, DIMD, DIMD, DIMD, DIMD);
  cvtw(prm(M2 + 0), wm2a, 512, DIMD, 512, DIMD);   cvtw(prm(M2 + 2), wm2b, DIMD, DIMD, DIMD, DIMD);
  cvtw(prm(MO + 0), wmoa, 512, DIMD, 512, DIMD);   cvtw(prm(MO + 2), wmob, DIMD, CLSD, DIMD, 128);

  auto gemm = [&](const _Float16* A, int lda, const _Float16* Bt, int ldb,
                  const float* bias, int relu,
                  _Float16* C16, int ldc16, int coloff,
                  float* C32, int ldc32, int M, int Npad, int Nvalid, int K) {
    dim3 grid(M / 32, Npad / 128), blk(256);
    gemm_wmma_kernel<<<grid, blk, 0, stream>>>(A, lda, Bt, ldb, bias, relu,
                                               C16, ldc16, coloff, C32, ldc32, Nvalid, K);
  };

  // one GIN conv (linear-first): y = A@W1; agg = scatter(y); t = relu(BN(y+agg+b1)); hcat_half = relu(t@W2+b2)
  auto run_conv = [&](int base, const _Float16* Ain, int lda, int K,
                      const _Float16* Wt1, int ldb1, const _Float16* Wt2,
                      const int* eidx, int coloff) {
    gemm(Ain, lda, Wt1, ldb1, nullptr, 0, nullptr, 0, 0, y, DIMD, NND, DIMD, DIMD, K);
    hipMemsetAsync(agg, 0, (size_t)NND * DIMD * 4, stream);
    {
      long long th = (long long)ED * 64;
      scatter_add_kernel<<<ceil_div(th, TB), TB, 0, stream>>>(y, eidx, ED, agg);
    }
    {
      int elems = NND * DIMD;
      bnrelu_kernel<<<ceil_div(elems, TB), TB, 0, stream>>>(
          y, agg, prm(base + 1), prm(base + 4), prm(base + 5), prm(base + 2), prm(base + 3),
          t16, elems);
    }
    gemm(t16, DIMD, Wt2, DIMD, prm(base + 7), 1, hcat16, 512, coloff, nullptr, 0,
         NND, DIMD, DIMD, DIMD);
  };

  // ---- block 1 (input features, K padded 1000->1024) ----
  run_conv(C11, x16, INP, INP, w11a, INP, w11b, e1, 0);
  run_conv(C12, x16, INP, INP, w12a, INP, w12b, e2, DIMD);
  gemm(hcat16, 512, wm1a, 512, prm(M1 + 1), 1, t16, DIMD, 0, nullptr, 0, NND, DIMD, DIMD, 512);
  gemm(t16, DIMD, wm1b, DIMD, prm(M1 + 3), 0, h16, DIMD, 0, nullptr, 0, NND, DIMD, DIMD, DIMD);

  // ---- block 2 ----
  run_conv(C21, h16, DIMD, DIMD, w21a, DIMD, w21b, e1, 0);
  run_conv(C22, h16, DIMD, DIMD, w22a, DIMD, w22b, e2, DIMD);
  gemm(hcat16, 512, wm2a, 512, prm(M2 + 1), 1, t16, DIMD, 0, nullptr, 0, NND, DIMD, DIMD, 512);
  gemm(t16, DIMD, wm2b, DIMD, prm(M2 + 3), 0, h16, DIMD, 0, nullptr, 0, NND, DIMD, DIMD, DIMD);

  // ---- block 3 (mlp_2 reused, as in source) ----
  run_conv(C31, h16, DIMD, DIMD, w31a, DIMD, w31b, e1, 0);
  run_conv(C32_, h16, DIMD, DIMD, w32a, DIMD, w32b, e2, DIMD);
  gemm(hcat16, 512, wm2a, 512, prm(M2 + 1), 1, t16, DIMD, 0, nullptr, 0, NND, DIMD, DIMD, 512);
  gemm(t16, DIMD, wm2b, DIMD, prm(M2 + 3), 0, h16, DIMD, 0, hf32, DIMD, NND, DIMD, DIMD, DIMD);

  // ---- pooling: g = [seg_mean(h, idx1) | seg_mean(h, idx2)]  [128, 512] ----
  hipMemsetAsync(gsum, 0, (size_t)BD * 512 * 4, stream);
  hipMemsetAsync(cnt, 0, (size_t)2 * BD * 4, stream);
  pool_cnt_kernel<<<ceil_div(NND, TB), TB, 0, stream>>>(idx1, idx2, cnt, NND);
  {
    long long th = (long long)NND * 64;
    pool_sum_kernel<<<ceil_div(th, TB), TB, 0, stream>>>(hf32, idx1, idx2, gsum, NND);
  }
  {
    int elems = BD * 512;
    pool_div_kernel<<<ceil_div(elems, TB), TB, 0, stream>>>(gsum, cnt, g16, elems);
  }

  // ---- output MLP + log_softmax ----
  gemm(g16, 512, wmoa, 512, prm(MO + 1), 1, p16, DIMD, 0, nullptr, 0, BD, DIMD, DIMD, 512);
  gemm(p16, DIMD, wmob, DIMD, prm(MO + 3), 0, nullptr, 0, 0, lgts, 128, BD, 128, CLSD, DIMD);
  logsoftmax_kernel<<<1, 128, 0, stream>>>(lgts, (float*)d_out);
}
